// ODPM_7619271983677
// MI455X (gfx1250) — matrix operand order repaired and checked by reading, not compiled, not run on hardware
//
#include <hip/hip_runtime.h>

// ---------------------------------------------------------------------------
// VMamba SS2D block for MI455X (gfx1250, wave32, WMMA + async-LDS streaming).
// B=2, DM=96, H=W=64, L=4096, DI=192, K=8, N=16, R=6, C_PROJ=38
// ---------------------------------------------------------------------------

#define BATCH 2
#define DM    96
#define HH    64
#define WW    64
#define LL    4096
#define DI    192
#define KDIR  8
#define NST   16
#define RRANK 6
#define CH    64            // scan chunk length (steps prefetched per buffer)

typedef __attribute__((ext_vector_type(16))) _Float16 v16h;
typedef __attribute__((ext_vector_type(8)))  float    v8f;

// ---- WMMA operand loaders (contiguous k order, same for A and B) -------------------
// A matrix 16x32 f16: lane m = lane&15; halves e -> K = hi*16 + e (16 contiguous halves)
__device__ __forceinline__ v16h load_a16(const _Float16* rowK, int hi) {
    v16h r;
    ((float4*)&r)[0] = *(const float4*)(rowK + hi * 16);
    ((float4*)&r)[1] = *(const float4*)(rowK + hi * 16 + 8);
    return r;
}
// B matrix 32x16 f16: lane n = lane&15; halves e -> K = hi*16 + e (16 contiguous halves)
__device__ __forceinline__ v16h load_b16(const _Float16* p16) {
    v16h r;
    ((float4*)&r)[0] = ((const float4*)p16)[0];
    ((float4*)&r)[1] = ((const float4*)p16)[1];
    return r;
}

// ---- CDNA5 async global->LDS (ASYNCcnt) ------------------------------------
__device__ __forceinline__ void async_b128(unsigned lds_off, const float* g) {
    asm volatile("global_load_async_to_lds_b128 %0, %1, off"
                 :: "v"(lds_off), "v"(g) : "memory");
}
__device__ __forceinline__ void wait_async_le2() {
    asm volatile("s_wait_asynccnt 0x2" ::: "memory");
}
__device__ __forceinline__ void wait_async_0() {
    asm volatile("s_wait_asynccnt 0x0" ::: "memory");
}

// ---- direction mapping: sequence index l (dir k) -> row-major pixel --------
__device__ __forceinline__ int dir_src(int k, int l) {
    int le = (k == 2 || k == 3 || k == 6 || k == 7) ? (LL - 1 - l) : l;
    int base = (k < 2) ? k : (k < 4) ? (k - 2) : (k < 6) ? k : (k - 2); // 0,1,0,1,4,5,4,5
    if (base == 0) return le;                       // row-major
    if (base == 1) {                                // transpose flatten
        int i = le >> 6, j = le & 63;
        return (j << 6) + i;
    }
    int j = le >> 6, i = le & 63;
    if (base == 4) return (i << 6) + ((i + j) & 63);        // diagonal
    return (i << 6) + ((j - i) & 63);                       // anti-diagonal
}

__device__ __forceinline__ float silu(float x) { return x / (1.f + __expf(-x)); }

// ---------------------------------------------------------------------------
// Kernel 0: weight/activation conversion to f16, padding, y accumulator zero.
// ---------------------------------------------------------------------------
__global__ void k_prep(const float* __restrict__ x, const float* __restrict__ w_in,
                       const float* __restrict__ xpw, const float* __restrict__ w_out,
                       _Float16* __restrict__ win_h, _Float16* __restrict__ xT_h,
                       _Float16* __restrict__ wproj_h, _Float16* __restrict__ wout_h,
                       float* __restrict__ y_t) {
    int tid = blockIdx.x * blockDim.x + threadIdx.x;
    int stride = gridDim.x * blockDim.x;
    for (int i = tid; i < 2 * DI * DM; i += stride) win_h[i] = (_Float16)w_in[i];
    for (int i = tid; i < DM * DI; i += stride) wout_h[i] = (_Float16)w_out[i];
    for (int i = tid; i < KDIR * 48 * DI; i += stride) {
        int d = i % DI, rk = i / DI, row = rk % 48, k = rk / 48;
        wproj_h[i] = (row < 38) ? (_Float16)xpw[((size_t)k * 38 + row) * DI + d]
                                : (_Float16)0.f;
    }
    for (int i = tid; i < BATCH * LL * DM; i += stride) {
        int c = i % DM, pb = i / DM, p = pb % LL, b = pb / LL;
        xT_h[i] = (_Float16)x[((size_t)b * DM + c) * LL + p];
    }
    for (int i = tid; i < BATCH * LL * DI; i += stride) y_t[i] = 0.f;
}

// ---------------------------------------------------------------------------
// Kernel 1: in_proj GEMM  xz[p][m] = w_in(384x96) * x(96 x L) per batch.
// 4 waves/block, each wave a 16x16 D tile, K=96 = 3 WMMA steps.
// ---------------------------------------------------------------------------
__global__ __launch_bounds__(128) void k_gemm_in(const _Float16* __restrict__ winh,
                                                 const _Float16* __restrict__ xTh,
                                                 float* __restrict__ xz_t) {
    int b = blockIdx.z;
    int ptile = blockIdx.x * 16;
    int wave = threadIdx.x >> 5, lane = threadIdx.x & 31;
    int n = lane & 15, hi = lane >> 4;
    int mtile = (blockIdx.y * 4 + wave) * 16;
    const _Float16* arow = winh + (size_t)(mtile + n) * DM;
    const _Float16* brow = xTh + ((size_t)b * LL + ptile + n) * DM;
    v8f acc = {};
#pragma unroll
    for (int ks = 0; ks < 3; ++ks) {
        v16h a = load_a16(arow + ks * 32, hi);
        v16h bb = load_b16(brow + ks * 32 + hi * 16);
        acc = __builtin_amdgcn_wmma_f32_16x16x32_f16(false, a, false, bb,
                                                     (short)0, acc, false, false);
    }
    float* drow = xz_t + ((size_t)b * LL + ptile + n) * (2 * DI) + mtile + 8 * hi;
    *(float4*)drow = ((float4*)&acc)[0];
    *(float4*)(drow + 4) = ((float4*)&acc)[1];
}

// ---------------------------------------------------------------------------
// Kernel 2: depthwise 3x3 conv + bias + SiLU on first DI channels; SiLU(z).
// ---------------------------------------------------------------------------
__global__ __launch_bounds__(192) void k_conv_act(const float* __restrict__ xz_t,
                                                  const float* __restrict__ conv_w,
                                                  const float* __restrict__ conv_b,
                                                  float* __restrict__ xc_t,
                                                  _Float16* __restrict__ xc_h,
                                                  float* __restrict__ z_t) {
    int b = blockIdx.y, p = blockIdx.x, d = threadIdx.x;
    int i = p >> 6, j = p & 63;
    float acc = conv_b[d];
#pragma unroll
    for (int dy = -1; dy <= 1; ++dy)
#pragma unroll
        for (int dx = -1; dx <= 1; ++dx) {
            int ii = i + dy, jj = j + dx;
            if (ii >= 0 && ii < HH && jj >= 0 && jj < WW) {
                float w = conv_w[d * 9 + (dy + 1) * 3 + (dx + 1)];
                acc += w * xz_t[((size_t)b * LL + (ii << 6) + jj) * (2 * DI) + d];
            }
        }
    float xcv = silu(acc);
    size_t o = ((size_t)b * LL + p) * DI + d;
    xc_t[o] = xcv;
    xc_h[o] = (_Float16)xcv;
    float zv = xz_t[((size_t)b * LL + p) * (2 * DI) + DI + d];
    z_t[o] = silu(zv);
}

// ---------------------------------------------------------------------------
// Kernel 3: fused x_proj WMMA (38->48 padded rows x K=192, direction gather in
// the B-operand loads) + dt_proj (K=6, VALU) + softplus -> delta, B, C.
// ---------------------------------------------------------------------------
__global__ __launch_bounds__(128) void k_xproj(const _Float16* __restrict__ wprojh,
                                               const _Float16* __restrict__ xc_h,
                                               const float* __restrict__ dtw,
                                               const float* __restrict__ dtb,
                                               float* __restrict__ delta_t,
                                               float* __restrict__ Bm,
                                               float* __restrict__ Cm) {
    int b = blockIdx.z, k = blockIdx.y, ltile = blockIdx.x * 16;
    __shared__ float xd[48][16];
    int wave = threadIdx.x >> 5, lane = threadIdx.x & 31;
    int n = lane & 15, hi = lane >> 4;
    if (wave < 3) {
        int p = dir_src(k, ltile + n);
        const _Float16* brow = xc_h + ((size_t)b * LL + p) * DI;
        const _Float16* arow = wprojh + ((size_t)k * 48 + wave * 16 + n) * DI;
        v8f acc = {};
#pragma unroll
        for (int ks = 0; ks < 6; ++ks) {
            v16h a = load_a16(arow + ks * 32, hi);
            v16h bb = load_b16(brow + ks * 32 + hi * 16);
            acc = __builtin_amdgcn_wmma_f32_16x16x32_f16(false, a, false, bb,
                                                         (short)0, acc, false, false);
        }
#pragma unroll
        for (int v = 0; v < 8; ++v) xd[wave * 16 + 8 * hi + v][n] = acc[v];
    }
    __syncthreads();
    size_t base_bk = (size_t)(b * KDIR + k) * LL + ltile;
    // dt_proj (K=6) + softplus -> delta[b][k][l][d]
    for (int idx = threadIdx.x; idx < 16 * DI; idx += 128) {
        int d = idx % DI, nn = idx / DI;
        const float* wrow = dtw + ((size_t)k * DI + d) * RRANK;
        float s = dtb[k * DI + d];
#pragma unroll
        for (int r = 0; r < RRANK; ++r) s += wrow[r] * xd[r][nn];
        float sp = (s > 20.f) ? s : log1pf(__expf(s));
        delta_t[(base_bk + nn) * DI + d] = sp;
    }
    // B rows 6..21, C rows 22..37 -> [b][k][l][n]
    for (int idx = threadIdx.x; idx < 2 * 16 * NST; idx += 128) {
        int nn = idx & 15, r = (idx >> 4) & 15, isC = idx >> 8;
        float v = xd[RRANK + isC * NST + r][nn];
        float* dst = isC ? Cm : Bm;
        dst[(base_bk + nn) * NST + r] = v;
    }
}

// ---------------------------------------------------------------------------
// Kernel 4: selective scan with async-LDS double buffering of B/C tiles.
// Block = 256: threads 0..191 compute (d = channel, 16 states in VGPRs);
// all 256 threads issue exactly two b128 async transfers per 64-step chunk,
// so ASYNCcnt bookkeeping is uniform and `s_wait_asynccnt 2` leaves the
// next chunk's loads in flight behind the serial exp/FMA chain.
// Merged output (= cross_merge) accumulated via global_atomic_add_f32.
// ---------------------------------------------------------------------------
__global__ __launch_bounds__(256) void k_scan(const float* __restrict__ delta_t,
                                              const float* __restrict__ xc_t,
                                              const float* __restrict__ Bm,
                                              const float* __restrict__ Cm,
                                              const float* __restrict__ A_logs,
                                              const float* __restrict__ Ds,
                                              float* __restrict__ y_t) {
    __shared__ float Bs[2][CH][NST];
    __shared__ float Cs[2][CH][NST];
    int k = blockIdx.x, b = blockIdx.y;
    int tid = threadIdx.x;
    size_t bk = (size_t)(b * KDIR + k) * LL;

    unsigned bs_base = (unsigned)(size_t)(&Bs[0][0][0]);
    unsigned cs_base = (unsigned)(size_t)(&Cs[0][0][0]);

    // thread t moves b128 #t of the B tile and #t of the C tile (256 each)
    auto issue_chunk = [&](int c, int buf) {
        const float* bsrc = Bm + (bk + (size_t)c * CH) * NST + tid * 4;
        const float* csrc = Cm + (bk + (size_t)c * CH) * NST + tid * 4;
        unsigned bdst = bs_base + (unsigned)buf * (CH * NST * 4) + tid * 16;
        unsigned cdst = cs_base + (unsigned)buf * (CH * NST * 4) + tid * 16;
        async_b128(bdst, bsrc);
        async_b128(cdst, csrc);
    };

    float A1[NST], h[NST], Dv = 0.f;
    int d = tid;
    if (d < DI) {
        const float* al = A_logs + ((size_t)k * DI + d) * NST;
#pragma unroll
        for (int nidx = 0; nidx < NST; ++nidx) {
            A1[nidx] = -__expf(al[nidx]);
            h[nidx] = 0.f;
        }
        Dv = Ds[k * DI + d];
    }

    issue_chunk(0, 0);
    for (int c = 0; c < LL / CH; ++c) {
        int buf = c & 1;
        if (c + 1 < LL / CH) {
            issue_chunk(c + 1, buf ^ 1);
            wait_async_le2();            // chunk c resident, c+1 in flight
        } else {
            wait_async_0();
        }
        __syncthreads();
        if (d < DI) {
            for (int t = 0; t < CH; ++t) {
                int l = c * CH + t;
                float delta = delta_t[(bk + l) * DI + d];
                if ((t & 15) == 0) {     // pull delta stream 2 chunks ahead
                    int lp = l + 2 * CH;
                    if (lp < LL) __builtin_prefetch(&delta_t[(bk + lp) * DI + d], 0, 0);
                }
                int p = dir_src(k, l);
                float u = xc_t[((size_t)b * LL + p) * DI + d];
                float du = delta * u;
                float acc = 0.f;
#pragma unroll
                for (int nidx = 0; nidx < NST; ++nidx) {
                    float e = __expf(delta * A1[nidx]);   // v_exp_f32
                    h[nidx] = fmaf(h[nidx], e, du * Bs[buf][t][nidx]);
                    acc = fmaf(h[nidx], Cs[buf][t][nidx], acc);
                }
                atomicAdd(&y_t[((size_t)b * LL + p) * DI + d], acc + u * Dv);
            }
        }
        __syncthreads();                 // buffer `buf` free for chunk c+2
    }
}

// ---------------------------------------------------------------------------
// Kernel 5: LayerNorm over d (192) + multiply SiLU(z), emit f16 for out GEMM.
// ---------------------------------------------------------------------------
__global__ __launch_bounds__(192) void k_ln(const float* __restrict__ y_t,
                                            const float* __restrict__ z_t,
                                            const float* __restrict__ lnw,
                                            const float* __restrict__ lnb,
                                            _Float16* __restrict__ yf_h) {
    __shared__ float red[8];
    int b = blockIdx.y, p = blockIdx.x, d = threadIdx.x;
    int wave = d >> 5, lane = d & 31;
    size_t o = ((size_t)b * LL + p) * DI;
    float v = y_t[o + d];
    float s = v;
#pragma unroll
    for (int off = 16; off; off >>= 1) s += __shfl_xor(s, off, 32);
    if (lane == 0) red[wave] = s;
    __syncthreads();
    if (threadIdx.x == 0) {
        float t = 0.f;
        for (int w = 0; w < 6; ++w) t += red[w];
        red[6] = t * (1.f / (float)DI);
    }
    __syncthreads();
    float mu = red[6];
    float dv = v - mu;
    float s2 = dv * dv;
#pragma unroll
    for (int off = 16; off; off >>= 1) s2 += __shfl_xor(s2, off, 32);
    __syncthreads();
    if (lane == 0) red[wave] = s2;
    __syncthreads();
    if (threadIdx.x == 0) {
        float t = 0.f;
        for (int w = 0; w < 6; ++w) t += red[w];
        red[7] = rsqrtf(t * (1.f / (float)DI) + 1e-5f);
    }
    __syncthreads();
    float rstd = red[7];
    float yn = (dv * rstd * lnw[d] + lnb[d]) * z_t[o + d];
    yf_h[o + d] = (_Float16)yn;
}

// ---------------------------------------------------------------------------
// Kernel 6: out_proj GEMM  out[b][o][p] = w_out(96x192) * yf(192 x L).
// ---------------------------------------------------------------------------
__global__ __launch_bounds__(192) void k_gemm_out(const _Float16* __restrict__ wouth,
                                                  const _Float16* __restrict__ yfh,
                                                  float* __restrict__ out) {
    int b = blockIdx.y;
    int ptile = blockIdx.x * 16;
    int wave = threadIdx.x >> 5, lane = threadIdx.x & 31;
    int n = lane & 15, hi = lane >> 4;
    int mtile = wave * 16;
    const _Float16* arow = wouth + (size_t)(mtile + n) * DI;
    const _Float16* brow = yfh + ((size_t)b * LL + ptile + n) * DI;
    v8f acc = {};
#pragma unroll
    for (int ks = 0; ks < 6; ++ks) {
        v16h a = load_a16(arow + ks * 32, hi);
        v16h bb = load_b16(brow + ks * 32 + hi * 16);
        acc = __builtin_amdgcn_wmma_f32_16x16x32_f16(false, a, false, bb,
                                                     (short)0, acc, false, false);
    }
#pragma unroll
    for (int v = 0; v < 8; ++v) {
        int oo = mtile + 8 * hi + v;
        out[((size_t)b * DM + oo) * LL + ptile + n] = acc[v];
    }
}

// ---------------------------------------------------------------------------
extern "C" void kernel_launch(void* const* d_in, const int* in_sizes, int n_in,
                              void* d_out, int out_size, void* d_ws, size_t ws_size,
                              hipStream_t stream) {
    const float* x    = (const float*)d_in[0];
    const float* w_in = (const float*)d_in[1];
    const float* cw   = (const float*)d_in[2];
    const float* cb   = (const float*)d_in[3];
    const float* xpw  = (const float*)d_in[4];
    const float* dtw  = (const float*)d_in[5];
    const float* dtb  = (const float*)d_in[6];
    const float* alog = (const float*)d_in[7];
    const float* ds   = (const float*)d_in[8];
    const float* lnw  = (const float*)d_in[9];
    const float* lnb  = (const float*)d_in[10];
    const float* wout = (const float*)d_in[11];
    float* out = (float*)d_out;

    char* ws = (char*)d_ws;
    size_t off = 0;
    auto take = [&](size_t bytes) {
        char* p = ws + off;
        off += (bytes + 255) & ~(size_t)255;
        return p;
    };
    _Float16* win_h   = (_Float16*)take((size_t)2 * DI * DM * 2);        // 384x96 f16
    _Float16* wproj_h = (_Float16*)take((size_t)KDIR * 48 * DI * 2);     // padded 48x192
    _Float16* wout_h  = (_Float16*)take((size_t)DM * DI * 2);
    _Float16* xT_h    = (_Float16*)take((size_t)BATCH * LL * DM * 2);
    _Float16* xc_h    = (_Float16*)take((size_t)BATCH * LL * DI * 2);
    float*    xz_t    = (float*)take((size_t)BATCH * LL * 2 * DI * 4);
    float*    xc_t    = (float*)take((size_t)BATCH * LL * DI * 4);
    float*    z_t     = (float*)take((size_t)BATCH * LL * DI * 4);
    float*    delta_t = (float*)take((size_t)BATCH * KDIR * LL * DI * 4);
    float*    Bm      = (float*)take((size_t)BATCH * KDIR * LL * NST * 4);
    float*    Cm      = (float*)take((size_t)BATCH * KDIR * LL * NST * 4);
    float*    y_t     = (float*)take((size_t)BATCH * LL * DI * 4);
    _Float16* yf_h    = (_Float16*)take((size_t)BATCH * LL * DI * 2);

    k_prep<<<1024, 256, 0, stream>>>(x, w_in, xpw, wout, win_h, xT_h, wproj_h,
                                     wout_h, y_t);
    k_gemm_in<<<dim3(LL / 16, 6, BATCH), 128, 0, stream>>>(win_h, xT_h, xz_t);
    k_conv_act<<<dim3(LL, BATCH), DI, 0, stream>>>(xz_t, cw, cb, xc_t, xc_h, z_t);
    k_xproj<<<dim3(LL / 16, KDIR, BATCH), 128, 0, stream>>>(wproj_h, xc_h, dtw, dtb,
                                                            delta_t, Bm, Cm);
    k_scan<<<dim3(KDIR, BATCH), 256, 0, stream>>>(delta_t, xc_t, Bm, Cm, alog, ds, y_t);
    k_ln<<<dim3(LL, BATCH), DI, 0, stream>>>(y_t, z_t, lnw, lnb, yf_h);
    k_gemm_out<<<dim3(LL / 16, BATCH), DI, 0, stream>>>(wout_h, yf_h, out);
}
